// TransformerXL_63015760167570
// MI455X (gfx1250) — matrix operand order repaired and checked by reading, not compiled, not run on hardware
//
#include <hip/hip_runtime.h>
#include <hip/hip_bf16.h>
#include <math.h>

// ---------------------------------------------------------------------------
// Transformer-XL forward, MI455X (gfx1250), bf16 WMMA + f32 accumulate.
// Async global->LDS staging (ASYNCcnt) with double-buffered tiles.
// ---------------------------------------------------------------------------

typedef __attribute__((ext_vector_type(16))) __bf16 v16bf;
typedef __attribute__((ext_vector_type(8)))  float  v8f;

#define CV   32000   // vocab
#define CD   1024    // model dim
#define CH   16      // heads
#define CNL  4       // layers
#define CDFF 4096
#define CDH  64      // head dim
#define CB   4
#define CT   1024
#define CBT  (CB*CT)

#define BKT   64     // K per staged tile (2 WMMA K-steps)
#define LDSTR 72     // LDS row stride (BKT + 8 pad), rows stay 16B aligned

enum { EPI_STORE = 0, EPI_BIAS, EPI_BIAS_RELU, EPI_RES, EPI_BIAS_RES, EPI_BDSHIFT };

struct GP {
    const __bf16* A;    // [M,K] row-major (bf16)
    const __bf16* B;    // TRANSB: [N,K] row-major ; else [K,N] row-major
    float*        C;    // f32 out (may be null unless EPI needs it)
    __bf16*       Cbf;  // optional bf16 mirror of C
    const float*  bias; // [N] or null
    const float*  res;  // residual, same layout/offsets as C
    int M, N, K, lda, ldb, ldc, innerB;
    long long sAo, sAi, sBo, sBi, sCo, sCi;
};

// Async copy of 16 bytes/lane from global to LDS (gfx1250 async path, ASYNCcnt).
__device__ __forceinline__ void async_ld16(void* lds, const void* g) {
    unsigned off = (unsigned)(size_t)lds;   // low 32 bits of flat addr == LDS offset
    asm volatile("global_load_async_to_lds_b128 %0, %1, off"
                 :: "v"(off), "v"(g) : "memory");
}

// Block tile 128x128, K-step 64 (two v_wmma_f32_16x16x32_bf16 per 16x16 tile).
// 256 threads = 8 waves; waves arranged 4 (M) x 2 (N); wave tile 32x64 = 2x4 WMMAs.
template<bool TRANSB, int EPI>
__global__ __launch_bounds__(256) void k_gemm(GP p) {
    __shared__ __bf16 As[2][128 * LDSTR];   // [m][k]
    __shared__ __bf16 Bs[2][128 * LDSTR];   // [n][k]

    const int tid  = threadIdx.x;
    const int lane = tid & 31;
    const int wid  = tid >> 5;
    const int wm   = wid & 3;          // 0..3  (M direction, 32 rows each)
    const int wn   = wid >> 2;         // 0..1  (N direction, 64 cols each)
    const int half = lane >> 4;        // 0/1
    const int lm   = lane & 15;

    const int bm = blockIdx.y * 128;
    const int bn = blockIdx.x * 128;

    const int zo = (int)blockIdx.z / p.innerB;
    const int zi = (int)blockIdx.z % p.innerB;
    const __bf16* Ag = p.A + (long long)zo * p.sAo + (long long)zi * p.sAi;
    const __bf16* Bg = p.B + (long long)zo * p.sBo + (long long)zi * p.sBi;
    float*  Cg  = p.C   ? p.C   + (long long)zo * p.sCo + (long long)zi * p.sCi : nullptr;
    __bf16* Cbg = p.Cbf ? p.Cbf + (long long)zo * p.sCo + (long long)zi * p.sCi : nullptr;
    const float* Rg = p.res ? p.res + (long long)zo * p.sCo + (long long)zi * p.sCi : nullptr;

    v8f acc[2][4] = {};

    // ---- tile staging ------------------------------------------------------
    // A tile 128x64 bf16 = 16KB = 1024 x 16B chunks -> 4 async per wave.
    auto stageA = [&](int kb, int buf) {
        const int k0 = kb * BKT;
        #pragma unroll
        for (int g = 0; g < 4; ++g) {
            int c   = tid + g * 256;
            int row = c >> 3;              // 8 chunks per row
            int ko  = (c & 7) * 8;
            async_ld16(&As[buf][row * LDSTR + ko],
                       Ag + (long long)(bm + row) * p.lda + k0 + ko);
        }
    };
    // B tile (TRANSB: [N,K] rows contiguous in K) -> pure async.
    // All TRANSB calls have N % 128 == 0, so no range guard needed.
    auto stageBt = [&](int kb, int buf) {
        const int k0 = kb * BKT;
        #pragma unroll
        for (int g = 0; g < 4; ++g) {
            int c   = tid + g * 256;
            int row = c >> 3;
            int ko  = (c & 7) * 8;
            async_ld16(&Bs[buf][row * LDSTR + ko],
                       Bg + (long long)(bn + row) * p.ldb + k0 + ko);
        }
    };
    // B tile (!TRANSB: [K,N]) -> transpose during staging via ds stores (DScnt).
    auto stageBn = [&](int kb, int buf) {
        const int k0 = kb * BKT;
        #pragma unroll
        for (int g = 0; g < 4; ++g) {
            int e    = tid + g * 256;      // 1024 groups of 8 n-contiguous
            int krow = e >> 4;             // 0..63
            int n0   = (e & 15) * 8;
            const __bf16* src = Bg + (long long)(k0 + krow) * p.ldb + bn + n0;
            #pragma unroll
            for (int j = 0; j < 8; ++j) {
                __bf16 v = (bn + n0 + j < p.N) ? src[j] : (__bf16)0.0f;
                Bs[buf][(n0 + j) * LDSTR + krow] = v;
            }
        }
    };
    auto stage = [&](int kb, int buf) {
        stageA(kb, buf);
        if (TRANSB) stageBt(kb, buf); else stageBn(kb, buf);
    };

    const int nK = p.K / BKT;
    stage(0, 0);

    for (int kb = 0; kb < nK; ++kb) {
        const int buf = kb & 1;
        if (kb + 1 < nK) {
            stage(kb + 1, buf ^ 1);
            // current tile's async loads complete when count drops to the
            // next tile's in-flight loads (in-order completion).
            if (TRANSB) asm volatile("s_wait_asynccnt 8" ::: "memory");
            else        asm volatile("s_wait_asynccnt 4" ::: "memory");
        } else {
            asm volatile("s_wait_asynccnt 0" ::: "memory");
        }
        __syncthreads();   // publish LDS tile across waves (also waits DScnt)

        // ---- 2 K-steps x 8 WMMAs ------------------------------------------
        #pragma unroll
        for (int kk = 0; kk < 2; ++kk) {
            v16bf af[2], bfr[4];
            #pragma unroll
            for (int tm = 0; tm < 2; ++tm) {
                const __bf16* base = &As[buf][(wm * 32 + tm * 16 + lm) * LDSTR + kk * 32];
                uint4* q = (uint4*)&af[tm];
                q[0] = *(const uint4*)(base + half * 8);        // K = half*8 .. +7
                q[1] = *(const uint4*)(base + 16 + half * 8);   // K = 16+half*8 .. +7
            }
            #pragma unroll
            for (int tn = 0; tn < 4; ++tn) {
                const __bf16* base = &Bs[buf][(wn * 64 + tn * 16 + lm) * LDSTR + kk * 32];
                uint4* q = (uint4*)&bfr[tn];
                q[0] = *(const uint4*)(base + half * 8);        // K = half*8 .. +7
                q[1] = *(const uint4*)(base + 16 + half * 8);   // K = 16+half*8 .. +7
            }
            #pragma unroll
            for (int tm = 0; tm < 2; ++tm)
                #pragma unroll
                for (int tn = 0; tn < 4; ++tn)
                    acc[tm][tn] = __builtin_amdgcn_wmma_f32_16x16x32_bf16(
                        false, af[tm], false, bfr[tn], (short)0, acc[tm][tn], false, false);
        }
        __syncthreads();   // all waves done reading buf before it is re-staged
    }

    // ---- epilogue: C layout: lanes 0-15 M=r, lanes 16-31 M=8+r ; N = lane%16
    #pragma unroll
    for (int tm = 0; tm < 2; ++tm) {
        #pragma unroll
        for (int tn = 0; tn < 4; ++tn) {
            #pragma unroll
            for (int r = 0; r < 8; ++r) {
                int m = bm + wm * 32 + tm * 16 + half * 8 + r;
                int n = bn + wn * 64 + tn * 16 + lm;
                if (m >= p.M || n >= p.N) continue;
                float v = acc[tm][tn][r];
                long long idx = (long long)m * p.ldc + n;
                if (EPI == EPI_BIAS || EPI == EPI_BIAS_RELU || EPI == EPI_BIAS_RES)
                    v += p.bias[n];
                if (EPI == EPI_BIAS_RELU)
                    v = v > 0.0f ? v : 0.0f;
                if (EPI == EPI_RES || EPI == EPI_BIAS_RES)
                    v += Rg[idx];
                if (EPI == EPI_BDSHIFT) {
                    // C logical element (i=m, jr=n). shifted[i,j]=BD[i, j-i+T-1]
                    int j = n + m - (p.N - 1);
                    if (j >= 0 && j <= m)
                        Cg[(long long)m * p.ldc + j] += v;
                } else {
                    if (Cg)  Cg[idx]  = v;
                    if (Cbg) Cbg[idx] = (__bf16)v;
                }
            }
        }
    }
}

// ---------------------------------------------------------------------------
// Support kernels
// ---------------------------------------------------------------------------

__global__ __launch_bounds__(256) void k_embed(const int* __restrict__ x,
                                               const float* __restrict__ emb,
                                               float* __restrict__ h) {
    long long idx = (long long)blockIdx.x * 256 + threadIdx.x;  // BT*D
    int tok = x[idx >> 10];
    int col = (int)(idx & 1023);
    h[idx] = emb[(long long)tok * CD + col] * 32.0f;            // sqrt(1024)=32
}

__global__ __launch_bounds__(256) void k_relpos(__bf16* __restrict__ rp) {
    long long idx = (long long)blockIdx.x * 256 + threadIdx.x;  // T*D
    int row = (int)(idx >> 10);
    int col = (int)(idx & 1023);
    float pos = (float)(CT - 1 - row);                          // slice-first
    int   f   = (col < 512) ? col : col - 512;
    float invf = expf(-(float)(2 * f) * (9.210340371976184f / 1024.0f)); // ln(1e4)
    float ang  = pos * invf;
    rp[idx] = (__bf16)((col < 512) ? sinf(ang) : cosf(ang));
}

// LayerNorm over D=1024, unbiased std (ddof=1), eps outside sqrt
__global__ __launch_bounds__(256) void k_layernorm(const float* __restrict__ x,
                                                   const float* __restrict__ ga,
                                                   const float* __restrict__ gb,
                                                   __bf16* __restrict__ out) {
    __shared__ float red[256];
    const int row = blockIdx.x;
    const int tid = threadIdx.x;
    const float4 xv = *(const float4*)(x + (long long)row * CD + tid * 4);
    float s = xv.x + xv.y + xv.z + xv.w;
    red[tid] = s; __syncthreads();
    for (int o = 128; o > 0; o >>= 1) { if (tid < o) red[tid] += red[tid + o]; __syncthreads(); }
    float mean = red[0] * (1.0f / CD);
    __syncthreads();
    float d0 = xv.x - mean, d1 = xv.y - mean, d2 = xv.z - mean, d3 = xv.w - mean;
    red[tid] = d0*d0 + d1*d1 + d2*d2 + d3*d3; __syncthreads();
    for (int o = 128; o > 0; o >>= 1) { if (tid < o) red[tid] += red[tid + o]; __syncthreads(); }
    float inv = 1.0f / (sqrtf(red[0] * (1.0f / (CD - 1))) + 1e-6f);
    int c = tid * 4;
    __bf16* o4 = out + (long long)row * CD + c;
    o4[0] = (__bf16)(ga[c+0] * d0 * inv + gb[c+0]);
    o4[1] = (__bf16)(ga[c+1] * d1 * inv + gb[c+1]);
    o4[2] = (__bf16)(ga[c+2] * d2 * inv + gb[c+2]);
    o4[3] = (__bf16)(ga[c+3] * d3 * inv + gb[c+3]);
}

__global__ __launch_bounds__(256) void k_qadd(const float* __restrict__ q,
                                              const float* __restrict__ u,
                                              const float* __restrict__ vb,
                                              __bf16* __restrict__ qu,
                                              __bf16* __restrict__ qv) {
    long long idx = (long long)blockIdx.x * 256 + threadIdx.x;  // BT*D
    int col = (int)(idx & 1023);                                 // h*64+d
    float qq = q[idx];
    qu[idx] = (__bf16)(qq + u[col]);
    qv[idx] = (__bf16)(qq + vb[col]);
}

// causal-masked scaled softmax: S f32 [BH,T,T] -> P bf16
__global__ __launch_bounds__(256) void k_softmax(const float* __restrict__ S,
                                                 __bf16* __restrict__ P) {
    __shared__ float red[256];
    const int i   = blockIdx.x;
    const long long bh = blockIdx.y;
    const int tid = threadIdx.x;
    const float* row = S + (bh * CT + i) * (long long)CT;
    __bf16* prow     = P + (bh * CT + i) * (long long)CT;
    float vals[4];
    float mx = -3.4e38f;
    #pragma unroll
    for (int c = 0; c < 4; ++c) {
        int j = tid + c * 256;
        float v = (j <= i) ? row[j] * 0.125f : -3.4e38f;       // 1/sqrt(64)
        vals[c] = v; mx = fmaxf(mx, v);
    }
    red[tid] = mx; __syncthreads();
    for (int o = 128; o > 0; o >>= 1) { if (tid < o) red[tid] = fmaxf(red[tid], red[tid+o]); __syncthreads(); }
    float m = red[0];
    __syncthreads();
    float sum = 0.0f;
    #pragma unroll
    for (int c = 0; c < 4; ++c) {
        int j = tid + c * 256;
        float e = (j <= i) ? __expf(vals[c] - m) : 0.0f;
        vals[c] = e; sum += e;
    }
    red[tid] = sum; __syncthreads();
    for (int o = 128; o > 0; o >>= 1) { if (tid < o) red[tid] += red[tid+o]; __syncthreads(); }
    float inv = 1.0f / red[0];
    #pragma unroll
    for (int c = 0; c < 4; ++c) {
        int j = tid + c * 256;
        prow[j] = (__bf16)(vals[c] * inv);
    }
}

// transpose+convert: src f32 [R,Cc] -> dst bf16 [Cc,R], batched over z
__global__ __launch_bounds__(256) void k_cvt_t(const float* __restrict__ src,
                                               __bf16* __restrict__ dst,
                                               int R, int Cc) {
    __shared__ float tl[32][33];
    long long off = (long long)blockIdx.z * R * Cc;
    src += off; dst += off;
    int c0 = blockIdx.x * 32, r0 = blockIdx.y * 32;
    int tx = threadIdx.x & 31, ty = threadIdx.x >> 5;  // 32x8
    #pragma unroll
    for (int k = 0; k < 4; ++k) {
        int r = r0 + ty + k * 8, c = c0 + tx;
        tl[ty + k * 8][tx] = (r < R && c < Cc) ? src[(long long)r * Cc + c] : 0.0f;
    }
    __syncthreads();
    #pragma unroll
    for (int k = 0; k < 4; ++k) {
        int c = c0 + ty + k * 8, r = r0 + tx;
        if (c < Cc && r < R) dst[(long long)c * R + r] = (__bf16)tl[tx][ty + k * 8];
    }
}

// ---------------------------------------------------------------------------
// Host-side dispatch
// ---------------------------------------------------------------------------

static void gemm_launch(hipStream_t s, bool transB, int epi,
                        const __bf16* A, const __bf16* B, float* C, __bf16* Cbf,
                        const float* bias, const float* res,
                        int M, int N, int K, int lda, int ldb, int ldc,
                        int batch, int innerB,
                        long long sAo, long long sAi, long long sBo, long long sBi,
                        long long sCo, long long sCi) {
    GP p;
    p.A = A; p.B = B; p.C = C; p.Cbf = Cbf; p.bias = bias; p.res = res;
    p.M = M; p.N = N; p.K = K; p.lda = lda; p.ldb = ldb; p.ldc = ldc;
    p.innerB = innerB;
    p.sAo = sAo; p.sAi = sAi; p.sBo = sBo; p.sBi = sBi; p.sCo = sCo; p.sCi = sCi;
    dim3 g((N + 127) / 128, (M + 127) / 128, batch);
    if (transB) {
        switch (epi) {
        case EPI_STORE:     k_gemm<true, EPI_STORE    ><<<g, 256, 0, s>>>(p); break;
        case EPI_BIAS:      k_gemm<true, EPI_BIAS     ><<<g, 256, 0, s>>>(p); break;
        case EPI_BIAS_RELU: k_gemm<true, EPI_BIAS_RELU><<<g, 256, 0, s>>>(p); break;
        case EPI_RES:       k_gemm<true, EPI_RES      ><<<g, 256, 0, s>>>(p); break;
        case EPI_BIAS_RES:  k_gemm<true, EPI_BIAS_RES ><<<g, 256, 0, s>>>(p); break;
        case EPI_BDSHIFT:   k_gemm<true, EPI_BDSHIFT  ><<<g, 256, 0, s>>>(p); break;
        }
    } else {
        k_gemm<false, EPI_STORE><<<g, 256, 0, s>>>(p);
    }
}

extern "C" void kernel_launch(void* const* d_in, const int* in_sizes, int n_in,
                              void* d_out, int out_size, void* d_ws, size_t ws_size,
                              hipStream_t stream) {
    (void)in_sizes; (void)n_in; (void)out_size; (void)ws_size;

    const int*   x     = (const int*)  d_in[0];
    const float* emb   = (const float*)d_in[1];
    const float* Wq    = (const float*)d_in[2];
    const float* Wk    = (const float*)d_in[3];
    const float* Wv    = (const float*)d_in[4];
    const float* Wr    = (const float*)d_in[5];
    const float* Wo    = (const float*)d_in[6];
    const float* u     = (const float*)d_in[7];
    const float* vbp   = (const float*)d_in[8];
    const float* ln1_a = (const float*)d_in[9];
    const float* ln1_b = (const float*)d_in[10];
    const float* ln2_a = (const float*)d_in[11];
    const float* ln2_b = (const float*)d_in[12];
    const float* W1    = (const float*)d_in[13];
    const float* b1    = (const float*)d_in[14];
    const float* W2    = (const float*)d_in[15];
    const float* b2    = (const float*)d_in[16];
    const float* lnf_a = (const float*)d_in[17];
    const float* lnf_b = (const float*)d_in[18];
    const float* Wp    = (const float*)d_in[19];
    const float* bp    = (const float*)d_in[20];
    float* logits = (float*)d_out;

    // ---- workspace carve-up ------------------------------------------------
    char* w = (char*)d_ws;
    auto alloc = [&](size_t bytes) -> void* {
        void* p = (void*)w;
        w += (bytes + 255) & ~(size_t)255;
        return p;
    };
    float*  h      = (float*) alloc((size_t)CBT * CD * 4);
    float*  q      = (float*) alloc((size_t)CBT * CD * 4);
    __bf16* xn     = (__bf16*)alloc((size_t)CBT * CD * 2);
    __bf16* qu     = (__bf16*)alloc((size_t)CBT * CD * 2);
    __bf16* qv     = (__bf16*)alloc((size_t)CBT * CD * 2);
    __bf16* kbf    = (__bf16*)alloc((size_t)CBT * CD * 2);
    __bf16* vbf    = (__bf16*)alloc((size_t)CBT * CD * 2);
    __bf16* aout   = (__bf16*)alloc((size_t)CBT * CD * 2);
    __bf16* midbf  = (__bf16*)alloc((size_t)CBT * CDFF * 2);
    __bf16* rpos   = (__bf16*)alloc((size_t)CT * CD * 2);
    __bf16* rbf    = (__bf16*)alloc((size_t)CT * CD * 2);
    float*  S      = (float*) alloc((size_t)CB * CH * CT * CT * 4);
    __bf16* P      = (__bf16*)alloc((size_t)CB * CH * CT * CT * 2);
    __bf16* Wq_t   = (__bf16*)alloc((size_t)CNL * CD * CD * 2);
    __bf16* Wk_t   = (__bf16*)alloc((size_t)CNL * CD * CD * 2);
    __bf16* Wv_t   = (__bf16*)alloc((size_t)CNL * CD * CD * 2);
    __bf16* Wr_t   = (__bf16*)alloc((size_t)CNL * CD * CD * 2);
    __bf16* Wo_t   = (__bf16*)alloc((size_t)CNL * CD * CD * 2);
    __bf16* W1_t   = (__bf16*)alloc((size_t)CNL * CD * CDFF * 2);
    __bf16* W2_t   = (__bf16*)alloc((size_t)CNL * CDFF * CD * 2);
    __bf16* Wp_t   = (__bf16*)alloc((size_t)CD * CV * 2);

    // ---- one-time prep: weights -> transposed bf16, embed, rel-pos table ---
    auto cvt_t = [&](const float* src, __bf16* dst, int R, int Cc, int batch) {
        dim3 g(Cc / 32, R / 32, batch);
        k_cvt_t<<<g, 256, 0, stream>>>(src, dst, R, Cc);
    };
    cvt_t(Wq, Wq_t, CD, CD, CNL);
    cvt_t(Wk, Wk_t, CD, CD, CNL);
    cvt_t(Wv, Wv_t, CD, CD, CNL);
    cvt_t(Wr, Wr_t, CD, CD, CNL);
    cvt_t(Wo, Wo_t, CD, CD, CNL);
    cvt_t(W1, W1_t, CD, CDFF, CNL);
    cvt_t(W2, W2_t, CDFF, CD, CNL);
    cvt_t(Wp, Wp_t, CD, CV, 1);

    k_embed <<<(CBT * CD) / 256, 256, 0, stream>>>(x, emb, h);
    k_relpos<<<(CT  * CD) / 256, 256, 0, stream>>>(rpos);

    const long long sTD  = (long long)CT * CD;
    const long long sHTT = (long long)CH * CT * CT;
    const long long sTT  = (long long)CT * CT;

    for (int i = 0; i < CNL; ++i) {
        const long long oDD  = (long long)i * CD * CD;
        const long long oDF  = (long long)i * CD * CDFF;

        // LN1(h) -> xn (bf16)
        k_layernorm<<<CBT, 256, 0, stream>>>(h, ln1_a + i * CD, ln1_b + i * CD, xn);

        // q (f32), k/v (bf16 mirrors)
        gemm_launch(stream, true, EPI_STORE, xn, Wq_t + oDD, q, nullptr, nullptr, nullptr,
                    CBT, CD, CD, CD, CD, CD, 1, 1, 0,0, 0,0, 0,0);
        gemm_launch(stream, true, EPI_STORE, xn, Wk_t + oDD, nullptr, kbf, nullptr, nullptr,
                    CBT, CD, CD, CD, CD, CD, 1, 1, 0,0, 0,0, 0,0);
        gemm_launch(stream, true, EPI_STORE, xn, Wv_t + oDD, nullptr, vbf, nullptr, nullptr,
                    CBT, CD, CD, CD, CD, CD, 1, 1, 0,0, 0,0, 0,0);

        // r = rpos @ Wr (bf16)
        gemm_launch(stream, true, EPI_STORE, rpos, Wr_t + oDD, nullptr, rbf, nullptr, nullptr,
                    CT, CD, CD, CD, CD, CD, 1, 1, 0,0, 0,0, 0,0);

        // qu = bf16(q+u), qv = bf16(q+vb)
        k_qadd<<<(CBT * CD) / 256, 256, 0, stream>>>(q, u + i * CD, vbp + i * CD, qu, qv);

        // S = AC = (q+u) @ k^T, batched over (b,h)
        gemm_launch(stream, true, EPI_STORE, qu, kbf, S, nullptr, nullptr, nullptr,
                    CT, CT, CDH, CD, CD, CT,
                    CB * CH, CH, sTD, CDH, sTD, CDH, sHTT, sTT);
        // S += rel_shift((q+vb) @ r^T)
        gemm_launch(stream, true, EPI_BDSHIFT, qv, rbf, S, nullptr, nullptr, nullptr,
                    CT, CT, CDH, CD, CD, CT,
                    CB * CH, CH, sTD, CDH, 0, CDH, sHTT, sTT);

        // P = softmax(mask(S/8))
        { dim3 g(CT, CB * CH); k_softmax<<<g, 256, 0, stream>>>(S, P); }

        // aout[b,i,h*64+d] = P @ v   (N=64 per head)
        gemm_launch(stream, false, EPI_STORE, P, vbf, nullptr, aout, nullptr, nullptr,
                    CT, CDH, CT, CT, CD, CD,
                    CB * CH, CH, sHTT, sTT, sTD, CDH, sTD, CDH);

        // h += aout @ Wo
        gemm_launch(stream, true, EPI_RES, aout, Wo_t + oDD, h, nullptr, nullptr, h,
                    CBT, CD, CD, CD, CD, CD, 1, 1, 0,0, 0,0, 0,0);

        // FFN
        k_layernorm<<<CBT, 256, 0, stream>>>(h, ln2_a + i * CD, ln2_b + i * CD, xn);
        gemm_launch(stream, true, EPI_BIAS_RELU, xn, W1_t + oDF, nullptr, midbf,
                    b1 + i * CDFF, nullptr,
                    CBT, CDFF, CD, CD, CD, CDFF, 1, 1, 0,0, 0,0, 0,0);
        gemm_launch(stream, true, EPI_BIAS_RES, midbf, W2_t + oDF, h, nullptr,
                    b2 + i * CD, h,
                    CBT, CD, CDFF, CDFF, CDFF, CD, 1, 1, 0,0, 0,0, 0,0);
    }

    // final LN + LM head
    k_layernorm<<<CBT, 256, 0, stream>>>(h, lnf_a, lnf_b, xn);
    gemm_launch(stream, true, EPI_BIAS, xn, Wp_t, logits, nullptr, bp, nullptr,
                CBT, CV, CD, CD, CD, CV, 1, 1, 0,0, 0,0, 0,0);
}
